// LightGlue_58660663329385
// MI455X (gfx1250) — matrix operand order repaired and checked by reading, not compiled, not run on hardware
//
#include <hip/hip_runtime.h>
#include <hip/hip_bf16.h>
#include <math.h>

// ---------------- problem constants ----------------
#define L_   9
#define D_   256
#define H_   4
#define N_   1024
#define B2_  4
#define HD_  64
#define M_   (B2_*N_)      // 4096 rows of activations
#define MAXM_ 50
#define THR_  0.1f

typedef _Float16 f16;
typedef unsigned int u32;
typedef __attribute__((ext_vector_type(16))) _Float16 v16h;
typedef __attribute__((ext_vector_type(8)))  _Float16 v8h;
typedef __attribute__((ext_vector_type(8)))  float    v8f;
typedef __attribute__((ext_vector_type(4)))  unsigned int u32x4;
typedef __attribute__((ext_vector_type(4)))  int i32x4;
typedef __attribute__((ext_vector_type(8)))  int i32x8;

#define V8F_ZERO {0.f,0.f,0.f,0.f,0.f,0.f,0.f,0.f}

// A-matrix 16x32 f16 fragment: lane<16 holds K {0..15},
// lane>=16 holds K {16..31}; row = lane&15. Two 16B chunks per lane.
__device__ __forceinline__ v16h load_afrag(const f16* p, int hh) {
  union { v16h v; v8h h[2]; } u;
  u.h[0] = *(const v8h*)(p + 16*hh);
  u.h[1] = *(const v8h*)(p + 16*hh + 8);
  return u.v;
}
// B-matrix 32x16 f16 fragment: col = lane&15, halves = K 16*hh .. 16*hh+15
// (K-contiguous per lane group, cf. SWMMAC B layout). One 32B contiguous read.
__device__ __forceinline__ v16h load_bfrag(const f16* p, int hh) {
  union { v16h v; v8h h[2]; } u;
  u.h[0] = *(const v8h*)(p + 16*hh);
  u.h[1] = *(const v8h*)(p + 16*hh + 8);
  return u.v;
}

__device__ __forceinline__ float lsig(float x) {
  return fminf(x, 0.f) - log1pf(__expf(-fabsf(x)));
}

// ---------------- TDM weight-slab prefetch (CDNA5 Tensor Data Mover) --------
// Each wave DMAs a 16-row slab of the weight matrix into LDS via a 2D tensor
// descriptor (1-byte elements). Contents are discarded; the transfer pulls the
// slab through L2 ahead of the consuming WMMA GEMM.
__global__ __launch_bounds__(32)
void tdm_prefetch_kernel(const float* __restrict__ src, int totalRows,
                         int rowBytes, int pitchBytes)
{
#if __has_builtin(__builtin_amdgcn_tensor_load_to_lds)
  __shared__ alignas(128) char stage[16 * 2048];
  const int r0 = blockIdx.x * 16;
  int rows = totalRows - r0; if (rows > 16) rows = 16; if (rows < 0) rows = 0;
  const unsigned long long ga =
      (unsigned long long)(size_t)src + (unsigned long long)r0 * (u32)pitchBytes;
  const u32 lds = (u32)(size_t)(void*)stage;
  // D# group 0: count=1 (valid), lds_addr[63:32], global_addr[120:64], type=2
  u32x4 g0;
  g0[0] = 1u;
  g0[1] = lds;
  g0[2] = (u32)(ga & 0xffffffffull);
  g0[3] = (u32)((ga >> 32) & 0x01ffffffull) | (2u << 30);
  // D# group 1: wg_mask=0, data_size=1B, 2D tile rowBytes x rows
  const u32 d0 = (u32)rowBytes;          // tensor_dim0 (bytes)
  const u32 d1 = (u32)rows;              // tensor_dim1 (rows)
  const u32 st = (u32)pitchBytes;        // tensor_dim0_stride (bytes)
  i32x8 g1 = {0,0,0,0,0,0,0,0};
  g1[1] = (int)((d0 & 0xffffu) << 16);                            // dim0[15:0] @63:48
  g1[2] = (int)(((d0 >> 16) & 0xffffu) | ((d1 & 0xffffu) << 16)); // dim0 hi | dim1 lo
  g1[3] = (int)(((d1 >> 16) & 0xffffu) | ((d0 & 0xffffu) << 16)); // dim1 hi | tile_dim0
  g1[4] = (int)(d1 & 0xffffu);                                    // tile_dim1 (tile_dim2=0)
  g1[5] = (int)st;                                                // dim0_stride[31:0]
  g1[6] = 0;                                                      // stride hi, dim1_stride
  g1[7] = 0;
  i32x4 gz = {0,0,0,0};
#if defined(__clang_major__) && (__clang_major__ >= 23)
  i32x8 gz8 = {0,0,0,0,0,0,0,0};
  __builtin_amdgcn_tensor_load_to_lds(g0, g1, gz, gz, gz8, 0);
#else
  __builtin_amdgcn_tensor_load_to_lds(g0, g1, gz, gz, 0);
#endif
  __builtin_amdgcn_s_wait_tensorcnt(0);
#else
  const int r = blockIdx.x * 16 + (threadIdx.x & 15);
  if (r < totalRows) __builtin_prefetch((const char*)src + (size_t)r * pitchBytes, 0, 1);
#endif
}

// ---------------- generic WMMA GEMM: out = (A @ W^T + bias) * oscale (+res) ----
// A: MxK f16 row-major.  W: NoutxK f16 row-major.
// 4 waves/block; each wave computes a 32x64 tile (2 A-frags x 4 B-frags,
// 8 WMMAs per 32-wide K step) for ~2.6x better L2 arithmetic intensity.
__global__ __launch_bounds__(128)
void wmma_gemm(const f16* __restrict__ A, const f16* __restrict__ W,
               const float* __restrict__ bias,
               float* __restrict__ out, f16* __restrict__ out16,
               const float* __restrict__ res,
               int M, int Nout, int K, float oscale)
{
  const int wid  = threadIdx.x >> 5;
  const int lane = threadIdx.x & 31;
  const int hh = lane >> 4, rc = lane & 15;
  const int m0 = blockIdx.x * 32;
  const int n0 = blockIdx.y * 256 + wid * 64;
  if (n0 >= Nout) return;

  int ar0 = m0 + rc;      if (ar0 > M - 1) ar0 = M - 1;
  int ar1 = m0 + 16 + rc; if (ar1 > M - 1) ar1 = M - 1;
  const f16* ap0 = A + (size_t)ar0 * K;
  const f16* ap1 = A + (size_t)ar1 * K;
  const f16* wp[4];
#pragma unroll
  for (int t = 0; t < 4; ++t) {
    int wr = n0 + t*16 + rc; if (wr > Nout - 1) wr = Nout - 1;
    wp[t] = W + (size_t)wr * K;
  }

  const v8f zero = V8F_ZERO;
  v8f acc[2][4];
#pragma unroll
  for (int r = 0; r < 2; ++r)
#pragma unroll
    for (int t = 0; t < 4; ++t) acc[r][t] = zero;

  for (int k0 = 0; k0 < K; k0 += 32) {
    __builtin_prefetch(ap0 + k0 + 256, 0, 1);   // global_prefetch_b8
    __builtin_prefetch(wp[0] + k0 + 256, 0, 1);
    const v16h a0 = load_afrag(ap0 + k0, hh);
    const v16h a1 = load_afrag(ap1 + k0, hh);
#pragma unroll
    for (int t = 0; t < 4; ++t) {
      const v16h b = load_bfrag(wp[t] + k0, hh);
      acc[0][t] = __builtin_amdgcn_wmma_f32_16x16x32_f16(false, a0, false, b, (short)0, acc[0][t], false, false);
      acc[1][t] = __builtin_amdgcn_wmma_f32_16x16x32_f16(false, a1, false, b, (short)0, acc[1][t], false, false);
    }
  }

#pragma unroll
  for (int t = 0; t < 4; ++t) {
    const int col = n0 + t*16 + rc;
    if (col < Nout) {
      const float bv = bias ? bias[col] : 0.f;
#pragma unroll
      for (int r = 0; r < 2; ++r) {
#pragma unroll
        for (int i = 0; i < 8; ++i) {
          const int row = m0 + r*16 + i + 8*hh;
          if (row < M) {
            const size_t o = (size_t)row * Nout + col;
            float v = (acc[r][t][i] + bv) * oscale;
            if (res)   v += res[o];
            if (out)   out[o]   = v;
            if (out16) out16[o] = (f16)v;
          }
        }
      }
    }
  }
}

// ---------------- fused flash attention ----------------
// Q,K: (B,H,N,HD) f16   Vt: (B,H,HD,N) f16   ctx16: (B,N,D) f16
// 4 waves/block share one (b,h); K/V tiles staged in LDS once per 32-key block
// (4x less L2 traffic), fragments re-read via ds_load_b128; online softmax.
__global__ __launch_bounds__(128)
void attn_kernel(const f16* __restrict__ Q, const f16* __restrict__ K,
                 const f16* __restrict__ Vt, f16* __restrict__ ctx16,
                 int kv_xor, float scale)
{
  __shared__ alignas(16) f16 ktile[32*64];     // [key][hd]
  __shared__ alignas(16) f16 vtile[64*32];     // [hd][key]
  __shared__ alignas(16) f16 pshm[4][16*32];
  const int tid  = threadIdx.x;
  const int wid  = tid >> 5;
  const int lane = tid & 31;
  const int hh = lane >> 4, rc = lane & 15;
  const int tile = blockIdx.x * 4 + wid;       // 0..1023
  const int bh = tile >> 6;                    // same for all 4 waves (64%4==0)
  const int qt = tile & 63;
  const int b = bh >> 2, h = bh & 3;
  const int kvbh = ((b ^ kv_xor) * H_) + h;
  f16* ps = pshm[wid];

  const f16* qrow = Q + ((size_t)bh * N_ + qt*16 + rc) * HD_;
  const v16h aq0 = load_afrag(qrow, hh);        // K 0..31
  const v16h aq1 = load_afrag(qrow + 32, hh);   // K 32..63

  const f16* Kbh = K  + (size_t)kvbh * N_ * HD_;
  const f16* Vbh = Vt + (size_t)kvbh * HD_ * N_;

  const v8f zero = V8F_ZERO;
  v8f acc0 = zero, acc1 = zero, acc2 = zero, acc3 = zero;
  float mrow[8], lrow[8];
#pragma unroll
  for (int i = 0; i < 8; ++i) { mrow[i] = -1e30f; lrow[i] = 0.f; }

  for (int kb = 0; kb < N_; kb += 32) {
    // cooperative LDS stage: 256 16B chunks each for K and Vt, 2 per thread
#pragma unroll
    for (int it = 0; it < 2; ++it) {
      const int ch = tid + it*128;                       // 0..255
      const int kr = ch >> 3, kc = (ch & 7) * 8;         // 32 rows x 8 chunks
      *(v8h*)(ktile + kr*64 + kc) = *(const v8h*)(Kbh + (size_t)(kb + kr)*HD_ + kc);
      const int vr = ch >> 2, vc = (ch & 3) * 8;         // 64 rows x 4 chunks
      *(v8h*)(vtile + vr*32 + vc) = *(const v8h*)(Vbh + (size_t)vr*N_ + kb + vc);
    }
    __syncthreads();

    v8f s0 = zero, s1 = zero;
    s0 = __builtin_amdgcn_wmma_f32_16x16x32_f16(false, aq0, false, load_bfrag(ktile + rc*64,           hh), (short)0, s0, false, false);
    s0 = __builtin_amdgcn_wmma_f32_16x16x32_f16(false, aq1, false, load_bfrag(ktile + rc*64 + 32,      hh), (short)0, s0, false, false);
    s1 = __builtin_amdgcn_wmma_f32_16x16x32_f16(false, aq0, false, load_bfrag(ktile + (rc+16)*64,      hh), (short)0, s1, false, false);
    s1 = __builtin_amdgcn_wmma_f32_16x16x32_f16(false, aq1, false, load_bfrag(ktile + (rc+16)*64 + 32, hh), (short)0, s1, false, false);

#pragma unroll
    for (int i = 0; i < 8; ++i) {
      float a0 = s0[i] * scale, a1 = s1[i] * scale;
      float rm = fmaxf(a0, a1);                   // reduce over 16 key-cols
      rm = fmaxf(rm, __shfl_xor(rm, 1));
      rm = fmaxf(rm, __shfl_xor(rm, 2));
      rm = fmaxf(rm, __shfl_xor(rm, 4));
      rm = fmaxf(rm, __shfl_xor(rm, 8));
      const float mn = fmaxf(mrow[i], rm);
      const float al = __expf(mrow[i] - mn);
      const float p0 = __expf(a0 - mn), p1 = __expf(a1 - mn);
      float rsum = p0 + p1;
      rsum += __shfl_xor(rsum, 1); rsum += __shfl_xor(rsum, 2);
      rsum += __shfl_xor(rsum, 4); rsum += __shfl_xor(rsum, 8);
      lrow[i] = lrow[i] * al + rsum; mrow[i] = mn;
      acc0[i] *= al; acc1[i] *= al; acc2[i] *= al; acc3[i] *= al;
      ps[(i + 8*hh)*32 + rc]      = (f16)p0;      // C-layout -> LDS
      ps[(i + 8*hh)*32 + 16 + rc] = (f16)p1;
    }
    __syncthreads();
    const v16h apv = load_afrag(ps + rc*32, hh);  // ds_load_b128 re-fragment
    acc0 = __builtin_amdgcn_wmma_f32_16x16x32_f16(false, apv, false, load_bfrag(vtile + rc*32,        hh), (short)0, acc0, false, false);
    acc1 = __builtin_amdgcn_wmma_f32_16x16x32_f16(false, apv, false, load_bfrag(vtile + (16+rc)*32,   hh), (short)0, acc1, false, false);
    acc2 = __builtin_amdgcn_wmma_f32_16x16x32_f16(false, apv, false, load_bfrag(vtile + (32+rc)*32,   hh), (short)0, acc2, false, false);
    acc3 = __builtin_amdgcn_wmma_f32_16x16x32_f16(false, apv, false, load_bfrag(vtile + (48+rc)*32,   hh), (short)0, acc3, false, false);
    __syncthreads();
  }
#pragma unroll
  for (int i = 0; i < 8; ++i) {
    const int row = qt*16 + i + 8*hh;
    const size_t o = ((size_t)b * N_ + row) * D_ + h*HD_ + rc;
    const float inv = 1.f / lrow[i];
    ctx16[o]      = (f16)(acc0[i] * inv);
    ctx16[o + 16] = (f16)(acc1[i] * inv);
    ctx16[o + 32] = (f16)(acc2[i] * inv);
    ctx16[o + 48] = (f16)(acc3[i] * inv);
  }
}

// ---------------- small elementwise / reduction kernels ----------------
__global__ void posenc_kernel(const float* __restrict__ kpts, const float* __restrict__ Wr,
                              float* __restrict__ ec, float* __restrict__ es)
{
  int idx = blockIdx.x * blockDim.x + threadIdx.x;      // (b*N+n)*32 + j
  if (idx >= B2_*N_*32) return;
  const int j = idx & 31, bn = idx >> 5;
  const float x = kpts[bn*2+0], y = kpts[bn*2+1];
  const float p = x*Wr[j*2+0] + y*Wr[j*2+1];
  const float cv = __cosf(p), sv = __sinf(p);
#pragma unroll
  for (int h = 0; h < H_; ++h) {
    const int d = h*64 + 2*j;
    ec[(size_t)bn*D_ + d] = cv; ec[(size_t)bn*D_ + d + 1] = cv;
    es[(size_t)bn*D_ + d] = sv; es[(size_t)bn*D_ + d + 1] = sv;
  }
}

__global__ void pack_x_kernel(const float* __restrict__ src, float* __restrict__ x,
                              f16* __restrict__ x16, int n)
{
  int i = blockIdx.x * blockDim.x + threadIdx.x;
  if (i < n) { float v = src[i]; x[i] = v; x16[i] = (f16)v; }
}

__global__ void cast_f32_f16_kernel(const float* __restrict__ a, f16* __restrict__ o, int n)
{
  int i = blockIdx.x * blockDim.x + threadIdx.x;
  if (i < n) o[i] = (f16)a[i];
}

// qkv (B,N,768) -> rope'd q16/k16 (B,H,N,HD) + v16t (B,H,HD,N)
__global__ void rope_kernel(const float* __restrict__ qkv, const float* __restrict__ ec,
                            const float* __restrict__ es, f16* __restrict__ q16,
                            f16* __restrict__ k16, f16* __restrict__ v16t)
{
  int idx = blockIdx.x * blockDim.x + threadIdx.x;      // (bn,h,p)
  if (idx >= B2_*N_*H_*32) return;
  const int p = idx & 31, h = (idx >> 5) & 3, bn = idx >> 7;
  const int b = bn >> 10, n = bn & 1023;
  const int d0 = h*64 + 2*p;
  const size_t base = (size_t)bn*768 + (size_t)d0*3;
  const float q0 = qkv[base+0], k0 = qkv[base+1], v0 = qkv[base+2];
  const float q1 = qkv[base+3], k1 = qkv[base+4], v1 = qkv[base+5];
  const float c = ec[(size_t)bn*D_ + d0], s = es[(size_t)bn*D_ + d0];
  const size_t hm = ((size_t)(b*H_ + h) * N_ + n) * HD_ + 2*p;
  q16[hm]   = (f16)(c*q0 - s*q1);
  q16[hm+1] = (f16)(c*q1 + s*q0);
  k16[hm]   = (f16)(c*k0 - s*k1);
  k16[hm+1] = (f16)(c*k1 + s*k0);
  const size_t vt = ((size_t)(b*H_ + h) * HD_ + 2*p) * N_ + n;
  v16t[vt]      = (f16)v0;
  v16t[vt + N_] = (f16)v1;
}

// qkc/vvc (B,N,D) f32 -> qk16 (B,H,N,HD), v16t (B,H,HD,N)
__global__ void prep_cross_kernel(const float* __restrict__ qkc, const float* __restrict__ vvc,
                                  f16* __restrict__ qk16, f16* __restrict__ v16t)
{
  int idx = blockIdx.x * blockDim.x + threadIdx.x;
  if (idx >= B2_*N_*D_) return;
  const int d = idx & 255, bn = idx >> 8;
  const int b = bn >> 10, n = bn & 1023, h = d >> 6, dd = d & 63;
  qk16[((size_t)(b*H_+h)*N_ + n)*HD_ + dd] = (f16)qkc[idx];
  v16t[((size_t)(b*H_+h)*HD_ + dd)*N_ + n] = (f16)vvc[idx];
}

__global__ void concat_cast_kernel(const float* __restrict__ x, const float* __restrict__ msg,
                                   f16* __restrict__ cat16)
{
  int idx = blockIdx.x * blockDim.x + threadIdx.x;
  if (idx >= M_*2*D_) return;
  const int r = idx >> 9, c = idx & 511;
  const float v = (c < D_) ? x[(size_t)r*D_ + c] : msg[(size_t)r*D_ + (c - D_)];
  cat16[idx] = (f16)v;
}

__global__ __launch_bounds__(128)
void ln_gelu_kernel(const float* __restrict__ hin, const float* __restrict__ g,
                    const float* __restrict__ bb, f16* __restrict__ o16, int C)
{
  __shared__ float r1[128], r2[128];
  const int row = blockIdx.x, tid = threadIdx.x;
  const float* hr = hin + (size_t)row * C;
  float s = 0.f, s2 = 0.f;
  for (int c = tid; c < C; c += 128) { float v = hr[c]; s += v; s2 += v*v; }
  r1[tid] = s; r2[tid] = s2; __syncthreads();
  for (int st = 64; st > 0; st >>= 1) {
    if (tid < st) { r1[tid] += r1[tid+st]; r2[tid] += r2[tid+st]; }
    __syncthreads();
  }
  const float mean = r1[0] / C;
  const float var  = r2[0] / C - mean*mean;
  const float rinv = rsqrtf(var + 1e-5f);
  for (int c = tid; c < C; c += 128) {
    const float y = (hr[c] - mean) * rinv * g[c] + bb[c];
    o16[(size_t)row*C + c] = (f16)(0.5f * y * (1.f + erff(y * 0.70710678118f)));
  }
}

__global__ __launch_bounds__(128)
void lse_row_kernel(const float* __restrict__ sim, float* __restrict__ rs)
{
  __shared__ float red[128];
  const int id = blockIdx.x, tid = threadIdx.x;       // id = p*N + n
  const float* row = sim + (size_t)id * N_;
  float mx = -1e30f;
  for (int m = tid; m < N_; m += 128) mx = fmaxf(mx, row[m]);
  red[tid] = mx; __syncthreads();
  for (int st = 64; st > 0; st >>= 1) { if (tid < st) red[tid] = fmaxf(red[tid], red[tid+st]); __syncthreads(); }
  mx = red[0]; __syncthreads();
  float s = 0.f;
  for (int m = tid; m < N_; m += 128) s += __expf(row[m] - mx);
  red[tid] = s; __syncthreads();
  for (int st = 64; st > 0; st >>= 1) { if (tid < st) red[tid] += red[tid+st]; __syncthreads(); }
  if (tid == 0) rs[id] = mx + logf(red[0]);
}

__global__ __launch_bounds__(128)
void lse_col_kernel(const float* __restrict__ sim, float* __restrict__ cs)
{
  __shared__ float red[128];
  const int id = blockIdx.x, tid = threadIdx.x;       // id = p*N + m
  const int p = id >> 10, m = id & 1023;
  const float* base = sim + (size_t)p*N_*N_ + m;
  float mx = -1e30f;
  for (int n = tid; n < N_; n += 128) mx = fmaxf(mx, base[(size_t)n*N_]);
  red[tid] = mx; __syncthreads();
  for (int st = 64; st > 0; st >>= 1) { if (tid < st) red[tid] = fmaxf(red[tid], red[tid+st]); __syncthreads(); }
  mx = red[0]; __syncthreads();
  float s = 0.f;
  for (int n = tid; n < N_; n += 128) s += __expf(base[(size_t)n*N_] - mx);
  red[tid] = s; __syncthreads();
  for (int st = 64; st > 0; st >>= 1) { if (tid < st) red[tid] += red[tid+st]; __syncthreads(); }
  if (tid == 0) cs[id] = mx + logf(red[0]);
}

__global__ void scores_kernel(const float* __restrict__ sim, const float* __restrict__ rs,
                              const float* __restrict__ cs, const float* __restrict__ z,
                              float* __restrict__ out)
{
  int idx = blockIdx.x * blockDim.x + threadIdx.x;
  if (idx >= 2*N_*N_) return;
  const int p = (idx >= N_*N_) ? 1 : 0;
  const int rem = idx - p*N_*N_;
  const int n = rem >> 10, m = rem & 1023;
  out[idx] = 2.f*sim[idx] - rs[p*N_ + n] - cs[p*N_ + m]
           + lsig(z[(size_t)(2*p)*N_ + n]) + lsig(z[(size_t)(2*p + 1)*N_ + m]);
}

// argmax over columns for each row of scores[pair0]; ms = exp(max)
__global__ __launch_bounds__(128)
void argmax_row_kernel(const float* __restrict__ sc, int* __restrict__ m0, float* __restrict__ ms)
{
  __shared__ float sv[128]; __shared__ int si[128];
  const int n = blockIdx.x, tid = threadIdx.x;
  float bv = -1e30f; int bi = 0;
  for (int m = tid; m < N_; m += 128) {
    const float v = sc[(size_t)n*N_ + m];
    if (v > bv) { bv = v; bi = m; }
  }
  sv[tid] = bv; si[tid] = bi; __syncthreads();
  for (int st = 64; st > 0; st >>= 1) {
    if (tid < st) {
      if (sv[tid+st] > sv[tid] || (sv[tid+st] == sv[tid] && si[tid+st] < si[tid])) {
        sv[tid] = sv[tid+st]; si[tid] = si[tid+st];
      }
    }
    __syncthreads();
  }
  if (tid == 0) { m0[n] = si[0]; ms[n] = __expf(sv[0]); }
}

__global__ __launch_bounds__(128)
void argmax_col_kernel(const float* __restrict__ sc, int* __restrict__ m1)
{
  __shared__ float sv[128]; __shared__ int si[128];
  const int m = blockIdx.x, tid = threadIdx.x;
  float bv = -1e30f; int bi = 0;
  for (int n = tid; n < N_; n += 128) {
    const float v = sc[(size_t)n*N_ + m];
    if (v > bv) { bv = v; bi = n; }
  }
  sv[tid] = bv; si[tid] = bi; __syncthreads();
  for (int st = 64; st > 0; st >>= 1) {
    if (tid < st) {
      if (sv[tid+st] > sv[tid] || (sv[tid+st] == sv[tid] && si[tid+st] < si[tid])) {
        sv[tid] = sv[tid+st]; si[tid] = si[tid+st];
      }
    }
    __syncthreads();
  }
  if (tid == 0) m1[m] = si[0];
}

// mutual filter + iterative top-50 (matches jax.lax.top_k tie order)
__global__ void filter_kernel(const int* __restrict__ m0, const int* __restrict__ m1,
                              const float* __restrict__ ms, float* __restrict__ tail)
{
  __shared__ float masked[N_];
  if (threadIdx.x != 0 || blockIdx.x != 0) return;
  for (int n = 0; n < N_; ++n) {
    const bool mut = (m1[m0[n]] == n);
    const float v = ms[n];
    masked[n] = (v > THR_ && mut) ? v : 0.f;
  }
  for (int k = 0; k < MAXM_; ++k) {
    float bv = -1e30f; int bi = 0;
    for (int n = 0; n < N_; ++n)
      if (masked[n] > bv) { bv = masked[n]; bi = n; }
    tail[k*3 + 0] = 0.f;
    tail[k*3 + 1] = (float)bi;
    tail[k*3 + 2] = (float)m0[bi];
    tail[3*MAXM_ + k] = bv;
    masked[bi] = -2e30f;
  }
}

// ---------------- host orchestration ----------------
extern "C" void kernel_launch(void* const* d_in, const int* in_sizes, int n_in,
                              void* d_out, int out_size, void* d_ws, size_t ws_size,
                              hipStream_t stream)
{
  (void)in_sizes; (void)n_in; (void)out_size; (void)ws_size;
  const float* kpts    = (const float*)d_in[0];
  const float* desc    = (const float*)d_in[1];
  const float* Wr      = (const float*)d_in[2];
  const float* sWqkv_w = (const float*)d_in[3];
  const float* sWqkv_b = (const float*)d_in[4];
  const float* sOut_w  = (const float*)d_in[5];
  const float* sOut_b  = (const float*)d_in[6];
  const float* sF1_w   = (const float*)d_in[7];
  const float* sF1_b   = (const float*)d_in[8];
  const float* sLN_g   = (const float*)d_in[9];
  const float* sLN_b   = (const float*)d_in[10];
  const float* sF2_w   = (const float*)d_in[11];
  const float* sF2_b   = (const float*)d_in[12];
  const float* cQK_w   = (const float*)d_in[13];
  const float* cQK_b   = (const float*)d_in[14];
  const float* cV_w    = (const float*)d_in[15];
  const float* cV_b    = (const float*)d_in[16];
  const float* cOut_w  = (const float*)d_in[17];
  const float* cOut_b  = (const float*)d_in[18];
  const float* cF1_w   = (const float*)d_in[19];
  const float* cF1_b   = (const float*)d_in[20];
  const float* cLN_g   = (const float*)d_in[21];
  const float* cLN_b   = (const float*)d_in[22];
  const float* cF2_w   = (const float*)d_in[23];
  const float* cF2_b   = (const float*)d_in[24];
  const float* fp_w    = (const float*)d_in[25];
  const float* fp_b    = (const float*)d_in[26];
  const float* match_w = (const float*)d_in[27];
  const float* match_b = (const float*)d_in[28];

  char* ws = (char*)d_ws;
  size_t off = 0;
  auto alloc = [&](size_t bytes) -> char* {
    char* p = ws + off;
    off += (bytes + 255) & ~(size_t)255;
    return p;
  };
  float* enc_c = (float*)alloc((size_t)M_*D_*4);
  float* enc_s = (float*)alloc((size_t)M_*D_*4);
  float* x     = (float*)alloc((size_t)M_*D_*4);
  f16*   x16   = (f16*)  alloc((size_t)M_*D_*2);
  float* qkv   = (float*)alloc((size_t)M_*3*D_*4);
  f16*   q16   = (f16*)  alloc((size_t)M_*D_*2);
  f16*   k16   = (f16*)  alloc((size_t)M_*D_*2);
  f16*   v16t  = (f16*)  alloc((size_t)M_*D_*2);
  f16*   ctx16 = (f16*)  alloc((size_t)M_*D_*2);
  float* msg   = (float*)alloc((size_t)M_*D_*4);
  f16*   cat16 = (f16*)  alloc((size_t)M_*2*D_*2);
  float* ffh   = (float*)alloc((size_t)M_*2*D_*4);
  f16*   ffg16 = (f16*)  alloc((size_t)M_*2*D_*2);
  float* qkc   = (float*)alloc((size_t)M_*D_*4);
  float* vvc   = (float*)alloc((size_t)M_*D_*4);
  f16*   wbuf  = (f16*)  alloc((size_t)512*512*2);
  f16*   md16  = (f16*)  alloc((size_t)M_*D_*2);
  float* zbuf  = (float*)alloc((size_t)M_*4);
  float* sim   = (float*)alloc((size_t)2*N_*N_*4);
  float* rs    = (float*)alloc((size_t)2*N_*4);
  float* cs    = (float*)alloc((size_t)2*N_*4);
  int*   am0   = (int*)  alloc((size_t)N_*4);
  int*   am1   = (int*)  alloc((size_t)N_*4);
  float* msb   = (float*)alloc((size_t)N_*4);

  auto cast16 = [&](const float* src, f16* dst, int n) {
    cast_f32_f16_kernel<<<(n + 255)/256, 256, 0, stream>>>(src, dst, n);
  };
  auto gemm = [&](const f16* A, const f16* Wm, const float* bias, float* out,
                  f16* o16, const float* res, int M, int Nn, int K, float sc) {
    dim3 g((M + 31)/32, (Nn + 255)/256);
    wmma_gemm<<<g, 128, 0, stream>>>(A, Wm, bias, out, o16, res, M, Nn, K, sc);
  };
  auto tdm_prefetch = [&](const float* src, int rows, int rowFloats) {
    tdm_prefetch_kernel<<<(rows + 15)/16, 32, 0, stream>>>(
        src, rows, rowFloats*4, rowFloats*4);
  };

  float* outf = (float*)d_out;

  // positional encodings (reused by every layer)
  posenc_kernel<<<(B2_*N_*32 + 255)/256, 256, 0, stream>>>(kpts, Wr, enc_c, enc_s);
  pack_x_kernel<<<(M_*D_ + 255)/256, 256, 0, stream>>>(desc, x, x16, M_*D_);

  const float iscale = 0.125f;   // 1/sqrt(HD)
  for (int i = 0; i < L_; ++i) {
    // ---- self attention ----
    tdm_prefetch(sWqkv_w + (size_t)i*3*D_*D_, 3*D_, D_);    // TDM warms L2
    cast16(sWqkv_w + (size_t)i*3*D_*D_, wbuf, 3*D_*D_);
    gemm(x16, wbuf, sWqkv_b + (size_t)i*3*D_, qkv, nullptr, nullptr, M_, 3*D_, D_, 1.f);
    rope_kernel<<<(B2_*N_*H_*32 + 255)/256, 256, 0, stream>>>(qkv, enc_c, enc_s, q16, k16, v16t);
    attn_kernel<<<256, 128, 0, stream>>>(q16, k16, v16t, ctx16, 0, iscale);
    cast16(sOut_w + (size_t)i*D_*D_, wbuf, D_*D_);
    gemm(ctx16, wbuf, sOut_b + (size_t)i*D_, msg, nullptr, nullptr, M_, D_, D_, 1.f);
    concat_cast_kernel<<<(M_*2*D_ + 255)/256, 256, 0, stream>>>(x, msg, cat16);
    tdm_prefetch(sF1_w + (size_t)i*2*D_*2*D_, 2*D_, 2*D_);
    cast16(sF1_w + (size_t)i*2*D_*2*D_, wbuf, 2*D_*2*D_);
    gemm(cat16, wbuf, sF1_b + (size_t)i*2*D_, ffh, nullptr, nullptr, M_, 2*D_, 2*D_, 1.f);
    ln_gelu_kernel<<<M_, 128, 0, stream>>>(ffh, sLN_g + (size_t)i*2*D_, sLN_b + (size_t)i*2*D_, ffg16, 2*D_);
    cast16(sF2_w + (size_t)i*D_*2*D_, wbuf, D_*2*D_);
    gemm(ffg16, wbuf, sF2_b + (size_t)i*D_, x, x16, x, M_, D_, 2*D_, 1.f);   // residual

    // ---- cross attention ----
    cast16(cQK_w + (size_t)i*D_*D_, wbuf, D_*D_);
    gemm(x16, wbuf, cQK_b + (size_t)i*D_, qkc, nullptr, nullptr, M_, D_, D_, 1.f);
    cast16(cV_w + (size_t)i*D_*D_, wbuf, D_*D_);
    gemm(x16, wbuf, cV_b + (size_t)i*D_, vvc, nullptr, nullptr, M_, D_, D_, 1.f);
    prep_cross_kernel<<<(M_*D_ + 255)/256, 256, 0, stream>>>(qkc, vvc, q16, v16t);
    attn_kernel<<<256, 128, 0, stream>>>(q16, q16, v16t, ctx16, 1, iscale);  // keys/values from partner
    cast16(cOut_w + (size_t)i*D_*D_, wbuf, D_*D_);
    gemm(ctx16, wbuf, cOut_b + (size_t)i*D_, msg, nullptr, nullptr, M_, D_, D_, 1.f);
    concat_cast_kernel<<<(M_*2*D_ + 255)/256, 256, 0, stream>>>(x, msg, cat16);
    tdm_prefetch(cF1_w + (size_t)i*2*D_*2*D_, 2*D_, 2*D_);
    cast16(cF1_w + (size_t)i*2*D_*2*D_, wbuf, 2*D_*2*D_);
    gemm(cat16, wbuf, cF1_b + (size_t)i*2*D_, ffh, nullptr, nullptr, M_, 2*D_, 2*D_, 1.f);
    ln_gelu_kernel<<<M_, 128, 0, stream>>>(ffh, cLN_g + (size_t)i*2*D_, cLN_b + (size_t)i*2*D_, ffg16, 2*D_);
    cast16(cF2_w + (size_t)i*D_*2*D_, wbuf, D_*2*D_);
    gemm(ffg16, wbuf, cF2_b + (size_t)i*D_, x, x16, x, M_, D_, 2*D_, 1.f);
  }

  // ---- final projections ----
  cast16(fp_w, wbuf, D_*D_);
  gemm(x16, wbuf, fp_b, nullptr, md16, nullptr, M_, D_, D_, 0.25f);   // / D^0.25
  cast16(match_w, wbuf, D_);
  gemm(x16, wbuf, match_b, zbuf, nullptr, nullptr, M_, 1, D_, 1.f);

  // sim[p] = md[2p] @ md[2p+1]^T  (WMMA, K=256)
  for (int p = 0; p < 2; ++p)
    gemm(md16 + (size_t)(2*p)*N_*D_, md16 + (size_t)(2*p + 1)*N_*D_,
         nullptr, sim + (size_t)p*N_*N_, nullptr, nullptr, N_, N_, D_, 1.f);

  lse_row_kernel<<<2*N_, 128, 0, stream>>>(sim, rs);
  lse_col_kernel<<<2*N_, 128, 0, stream>>>(sim, cs);
  scores_kernel<<<(2*N_*N_ + 255)/256, 256, 0, stream>>>(sim, rs, cs, zbuf, outf);

  // ---- match filtering on pair 0 ----
  argmax_row_kernel<<<N_, 128, 0, stream>>>(outf, am0, msb);
  argmax_col_kernel<<<N_, 128, 0, stream>>>(outf, am1);
  filter_kernel<<<1, 32, 0, stream>>>(am0, am1, msb, outf + (size_t)2*N_*N_);
}
